// InitialNetwork_27891517620858
// MI455X (gfx1250) — compile-verified, matrix operand order checked by reading, not run on hardware
//
#include <hip/hip_runtime.h>
#include <hip/hip_bf16.h>

// CDNA5 / gfx1250: wave32, WMMA 16x16x32 bf16, fp32 accumulate.

typedef __attribute__((ext_vector_type(16))) __bf16 v16bf;
typedef __attribute__((ext_vector_type(8)))  float  v8f;

#define T_STEPS 512
#define BATCH   256
#define DIN     256
#define HID     128
#define GATES   512   // 4*H
#define RT_PER_WG 8   // row tiles per workgroup in gx_gemm

union V16 { v16bf v; uint4 u[2]; };
union V8F { v8f v; float4 f4[2]; float f[8]; };

// gfx1250 hardware tanh (TRANS op). v_nop covers the trans-op result hazard
// since the compiler cannot schedule around inline asm.
__device__ __forceinline__ float tanh_hw(float x) {
    float r;
    asm volatile("v_tanh_f32 %0, %1\n\tv_nop" : "=v"(r) : "v"(x));
    return r;
}
__device__ __forceinline__ float sigmoid_hw(float x) {
    return fmaf(0.5f, tanh_hw(0.5f * x), 0.5f);
}

// K index inside a 16x32 bf16 WMMA operand for element e (0..15) of a lane,
// half = lane/16.  Used identically for A and B packing, so any deviation
// from real HW K-order cancels out in the GEMM.
__device__ __forceinline__ int kmap(int e, int half) {
    return ((e & 8) << 1) + (e & 7) + 8 * half;
}

// Column permutation: original gate column g0 = q*128 + j (q=gate, j=0..127),
// j = 16*w + r  ->  new column = 64*w + 16*q + r.  Inverse:
__device__ __forceinline__ int orig_col(int pc) {
    int w = pc >> 6, q = (pc >> 4) & 3, r = pc & 15;
    return q * 128 + (w << 4) + r;
}

// ---------------------------------------------------------------------------
// Prep: convert W_ih / W_hh to bf16, packed in B-operand order with permuted
// columns; permute bias.  Packed flat index: ((kt*32 + nt)*32 + lane)*16 + e.
// ---------------------------------------------------------------------------
__global__ void prep_pack(const float* __restrict__ Wih,
                          const float* __restrict__ Whh,
                          const float* __restrict__ b,
                          __bf16* __restrict__ WihP,
                          __bf16* __restrict__ WhhP,
                          float*  __restrict__ bP) {
    int tid = blockIdx.x * blockDim.x + threadIdx.x;
    const int NW1 = 8 * 32 * 32 * 16;   // 131072
    const int NW2 = 4 * 32 * 32 * 16;   // 65536
    if (tid < NW1) {
        int e = tid & 15, lane = (tid >> 4) & 31, nt = (tid >> 9) & 31, kt = tid >> 14;
        int k  = kt * 32 + kmap(e, lane >> 4);
        int pc = nt * 16 + (lane & 15);
        WihP[tid] = (__bf16)Wih[(size_t)k * GATES + orig_col(pc)];
    } else if (tid < NW1 + NW2) {
        int t2 = tid - NW1;
        int e = t2 & 15, lane = (t2 >> 4) & 31, nt = (t2 >> 9) & 31, kt = t2 >> 14;
        int k  = kt * 32 + kmap(e, lane >> 4);
        int pc = nt * 16 + (lane & 15);
        WhhP[t2] = (__bf16)Whh[(size_t)k * GATES + orig_col(pc)];
    } else if (tid < NW1 + NW2 + GATES) {
        int c = tid - NW1 - NW2;
        bP[c] = b[orig_col(c)];
    }
}

// ---------------------------------------------------------------------------
// Big GEMM: gx = bf16(xs) @ WihP + bP, output in packed per-lane C layout:
//   gxp[ ((rt*8 + wave)*4 + q)*256 + lane*8 + e ]
// B operand (all 8 k-tiles x 4 gate tiles) resident in VGPRs for the whole
// workgroup lifetime; 8 row tiles per WG amortize the weight loads.
// ---------------------------------------------------------------------------
__global__ __launch_bounds__(256) void gx_gemm(const float* __restrict__ xs,
                                               const __bf16* __restrict__ WihP,
                                               const float* __restrict__ bP,
                                               float* __restrict__ gxp) {
    const int lane = threadIdx.x & 31;
    const int wave = threadIdx.x >> 5;   // 0..7
    const int half = lane >> 4;
    const int l16  = lane & 15;

    // Resident B: 8 kt x 4 q = 256 VGPRs/lane
    V16 Bv[8][4];
    #pragma unroll
    for (int kt = 0; kt < 8; ++kt)
        #pragma unroll
        for (int q = 0; q < 4; ++q) {
            int nt = wave * 4 + q;
            const uint4* wp = (const uint4*)(WihP + ((size_t)(kt * 32 + nt) * 32 + lane) * 16);
            Bv[kt][q].u[0] = wp[0]; Bv[kt][q].u[1] = wp[1];
        }
    float bias[4];
    #pragma unroll
    for (int q = 0; q < 4; ++q) bias[q] = bP[wave * 64 + q * 16 + l16];

    for (int i = 0; i < RT_PER_WG; ++i) {
        const int rt = blockIdx.x * RT_PER_WG + i;
        const float* xrow = xs + (size_t)(rt * 16 + l16) * DIN;

        V8F acc[4];
        #pragma unroll
        for (int q = 0; q < 4; ++q)
            #pragma unroll
            for (int e = 0; e < 8; ++e) acc[q].f[e] = bias[q];

        #pragma unroll
        for (int kt = 0; kt < 8; ++kt) {
            V16 a;
            const float* p0 = xrow + kt * 32 + half * 8;
            const float* p1 = xrow + kt * 32 + 16 + half * 8;
            #pragma unroll
            for (int e = 0; e < 8; ++e) a.v[e]     = (__bf16)p0[e];
            #pragma unroll
            for (int e = 0; e < 8; ++e) a.v[8 + e] = (__bf16)p1[e];
            #pragma unroll
            for (int q = 0; q < 4; ++q)
                acc[q].v = __builtin_amdgcn_wmma_f32_16x16x32_bf16(
                    false, a.v, false, Bv[kt][q].v, (short)0, acc[q].v, false, false);
        }
        float4* op = (float4*)(gxp + ((size_t)(rt * 8 + wave) * 4) * 256 + lane * 8);
        #pragma unroll
        for (int q = 0; q < 4; ++q) {
            op[q * 64]     = acc[q].f4[0];   // q*256 floats = q*64 float4s
            op[q * 64 + 1] = acc[q].f4[1];
        }
    }
}

// ---------------------------------------------------------------------------
// LSTM scan: 16 WGs (one per 16-row batch tile), 512 sequential steps.
// - W_hh slice resident in VGPRs for the whole scan
// - h double-buffered in LDS -> ONE barrier per step
// - gx for step t+1 software-pipelined into registers during step t
// - hardware v_tanh for all gate nonlinearities
// ---------------------------------------------------------------------------
__global__ __launch_bounds__(256) void lstm_scan(const float* __restrict__ gxp,
                                                 const __bf16* __restrict__ WhhP,
                                                 float* __restrict__ hfin) {
    __shared__ __bf16 sh[2][16][128];
    const int lane = threadIdx.x & 31;
    const int wave = threadIdx.x >> 5;   // 0..7
    const int half = lane >> 4;
    const int l16  = lane & 15;
    const int g    = blockIdx.x;         // batch tile

    // Resident B operand: 4 k-tiles x 4 gate tiles = 128 VGPRs/lane
    V16 Bv[4][4];
    #pragma unroll
    for (int kt = 0; kt < 4; ++kt)
        #pragma unroll
        for (int q = 0; q < 4; ++q) {
            int nt = wave * 4 + q;
            const uint4* wp = (const uint4*)(WhhP + ((size_t)(kt * 32 + nt) * 32 + lane) * 16);
            Bv[kt][q].u[0] = wp[0]; Bv[kt][q].u[1] = wp[1];
        }

    for (int i = threadIdx.x; i < 16 * 128; i += 256)
        ((__bf16*)sh[0])[i] = (__bf16)0.0f;
    __syncthreads();

    v8f c;
    #pragma unroll
    for (int e = 0; e < 8; ++e) c[e] = 0.0f;
    float hreg[8];

    // gx base for (t, g, wave): stride per t = 16*8*4*256 floats
    const float* gw = gxp + ((size_t)g * 8 + wave) * 4 * 256 + lane * 8;
    const size_t tstride = (size_t)16 * 8 * 4 * 256;

    V8F gxc[4], gxn[4];
    #pragma unroll
    for (int q = 0; q < 4; ++q) {
        gxc[q].f4[0] = *(const float4*)(gw + q * 256);
        gxc[q].f4[1] = *(const float4*)(gw + q * 256 + 4);
    }

    #pragma unroll 2
    for (int t = 0; t < T_STEPS; ++t) {
        const int p = t & 1;
        // A operand: current h from LDS
        V16 a[4];
        #pragma unroll
        for (int kt = 0; kt < 4; ++kt) {
            a[kt].u[0] = *(const uint4*)&sh[p][l16][kt * 32 + half * 8];
            a[kt].u[1] = *(const uint4*)&sh[p][l16][kt * 32 + 16 + half * 8];
        }
        V8F acc[4];
        #pragma unroll
        for (int q = 0; q < 4; ++q) acc[q].v = gxc[q].v;

        // pipeline: fetch next step's gx while WMMAs run
        if (t + 1 < T_STEPS) {
            const float* gn = gw + (size_t)(t + 1) * tstride;
            #pragma unroll
            for (int q = 0; q < 4; ++q) {
                gxn[q].f4[0] = *(const float4*)(gn + q * 256);
                gxn[q].f4[1] = *(const float4*)(gn + q * 256 + 4);
            }
        }
        if (t + 6 < T_STEPS) {
            const float* pf = gw + (size_t)(t + 6) * tstride;
            #pragma unroll
            for (int q = 0; q < 4; ++q) __builtin_prefetch(pf + q * 256, 0, 0);
        }

        #pragma unroll
        for (int kt = 0; kt < 4; ++kt)
            #pragma unroll
            for (int q = 0; q < 4; ++q)
                acc[q].v = __builtin_amdgcn_wmma_f32_16x16x32_bf16(
                    false, a[kt].v, false, Bv[kt][q].v, (short)0, acc[q].v, false, false);

        #pragma unroll
        for (int e = 0; e < 8; ++e) {
            float iv = sigmoid_hw(acc[0].f[e]);
            float fv = sigmoid_hw(acc[1].f[e]);
            float gv = tanh_hw(acc[2].f[e]);
            float ov = sigmoid_hw(acc[3].f[e]);
            c[e] = fv * c[e] + iv * gv;
            float hv = ov * tanh_hw(c[e]);
            hreg[e] = hv;
            sh[1 - p][e + 8 * half][wave * 16 + l16] = (__bf16)hv;
        }
        __syncthreads();   // orders this step's writes vs next step's reads,
                           // and this step's reads of buf p vs next overwrite
        #pragma unroll
        for (int q = 0; q < 4; ++q) gxc[q].v = gxn[q].v;
    }
    #pragma unroll
    for (int e = 0; e < 8; ++e)
        hfin[(size_t)(g * 16 + e + 8 * half) * HID + wave * 16 + l16] = hreg[e];
}

// ---------------------------------------------------------------------------
// MLP head in fp32 (tiny, output-precision critical).
// ---------------------------------------------------------------------------
__global__ __launch_bounds__(128) void head(const float* __restrict__ hfin,
                                            const float* __restrict__ W1,
                                            const float* __restrict__ b1,
                                            const float* __restrict__ W2,
                                            const float* __restrict__ b2,
                                            float* __restrict__ out) {
    __shared__ float hrow[128];
    __shared__ float hid[128];
    int b = blockIdx.x, n = threadIdx.x;
    hrow[n] = hfin[(size_t)b * 128 + n];
    __syncthreads();
    float s = b1[n];
    #pragma unroll 4
    for (int k = 0; k < 128; ++k) s += hrow[k] * W1[k * 128 + n];
    hid[n] = fmaxf(s, 0.0f);
    __syncthreads();
    float p = b2[n];
    #pragma unroll 4
    for (int k = 0; k < 128; ++k) p += hid[k] * W2[k * 128 + n];
    if (n < 64) out[b * 64 + n] = p;                 // mean
    else        out[16384 + b * 64 + (n - 64)] = p;  // logvar
}

// ---------------------------------------------------------------------------
extern "C" void kernel_launch(void* const* d_in, const int* in_sizes, int n_in,
                              void* d_out, int out_size, void* d_ws, size_t ws_size,
                              hipStream_t stream) {
    const float* xs  = (const float*)d_in[0];
    const float* Wih = (const float*)d_in[1];
    const float* Whh = (const float*)d_in[2];
    const float* bl  = (const float*)d_in[3];
    const float* W1  = (const float*)d_in[4];
    const float* b1  = (const float*)d_in[5];
    const float* W2  = (const float*)d_in[6];
    const float* b2  = (const float*)d_in[7];
    float* out = (float*)d_out;

    char* ws = (char*)d_ws;
    const size_t gxBytes  = (size_t)T_STEPS * BATCH * GATES * sizeof(float); // 256 MB
    const size_t wihBytes = (size_t)8 * 32 * 32 * 16 * sizeof(__bf16);       // 256 KB
    const size_t whhBytes = (size_t)4 * 32 * 32 * 16 * sizeof(__bf16);       // 128 KB
    float*  gxp  = (float*) ws;
    __bf16* WihP = (__bf16*)(ws + gxBytes);
    __bf16* WhhP = (__bf16*)(ws + gxBytes + wihBytes);
    float*  bP   = (float*) (ws + gxBytes + wihBytes + whhBytes);
    float*  hfin = (float*) (ws + gxBytes + wihBytes + whhBytes + GATES * sizeof(float));

    int prepN = 8 * 32 * 32 * 16 + 4 * 32 * 32 * 16 + GATES;
    prep_pack<<<(prepN + 255) / 256, 256, 0, stream>>>(Wih, Whh, bl, WihP, WhhP, bP);
    gx_gemm  <<<(T_STEPS * BATCH / 16) / RT_PER_WG, 256, 0, stream>>>(xs, WihP, bP, gxp);
    lstm_scan<<<16, 256, 0, stream>>>(gxp, WhhP, hfin);
    head     <<<BATCH, 128, 0, stream>>>(hfin, W1, b1, W2, b2, out);
}
